// MultiHeadAttentionLayer_29901562314792
// MI455X (gfx1250) — matrix operand order repaired and checked by reading, not compiled, not run on hardware
//
#include <hip/hip_runtime.h>

// ---------------------------------------------------------------------------
// MultiHeadAttention forward for MI455X (gfx1250, wave32, WMMA bf16 path)
// B=4, S=2048, H=1024, NH=16, HD=64
// d_out = [outputs f32 (B*S*H)] ++ [attention f32 (B*NH*S*S)]
// Workspace layout (~120 MB):
//   bf16 copies of query/key/value, bf16 Wq/Wk/Wv/Wo,
//   bf16 Qproj, Kproj, Vt (per-head transposed), Ctx
// ---------------------------------------------------------------------------

constexpr int Bc  = 4;
constexpr int Sc  = 2048;
constexpr int Hc  = 1024;
constexpr int NHc = 16;
constexpr int HDc = 64;
constexpr int Mtot = Bc * Sc;          // 8192 rows for the projection GEMMs
constexpr float kInvScale = 0.125f;    // 1/sqrt(HD)
constexpr float kNeg = -1e10f;

typedef __attribute__((ext_vector_type(16))) __bf16       bf16x16;
typedef __attribute__((ext_vector_type(8)))  float        f32x8;
typedef __attribute__((ext_vector_type(4)))  float        f32x4;
typedef __attribute__((ext_vector_type(4)))  unsigned int u32x4;
typedef __attribute__((ext_vector_type(2)))  unsigned int u32x2;

union Frag {                 // one WMMA A/B operand: 32 bytes per lane
  u32x4   u[2];
  bf16x16 b;
};

__device__ __forceinline__ unsigned short f2bf(float f) {
  unsigned int u = __float_as_uint(f);
  u += 0x7FFFu + ((u >> 16) & 1u);     // round-to-nearest-even
  return (unsigned short)(u >> 16);
}

__device__ __forceinline__ f32x8 wmma_bf16(bf16x16 a, bf16x16 b, f32x8 c) {
  // D = A(16x32) * B(32x16) + C, f32 accumulate
  return __builtin_amdgcn_wmma_f32_16x16x32_bf16(false, a, false, b,
                                                 (short)0, c, false, false);
}

// ---------------------------------------------------------------------------
// f32 -> bf16 bulk convert (vectorized: 4 elems / thread)
// ---------------------------------------------------------------------------
__global__ __launch_bounds__(256)
void cvt_bf16_kernel(const float* __restrict__ in,
                     unsigned short* __restrict__ out, int n4) {
  int i = blockIdx.x * blockDim.x + threadIdx.x;
  if (i >= n4) return;
  f32x4 f = reinterpret_cast<const f32x4*>(in)[i];
  u32x2 o;
  o.x = (unsigned int)f2bf(f.x) | ((unsigned int)f2bf(f.y) << 16);
  o.y = (unsigned int)f2bf(f.z) | ((unsigned int)f2bf(f.w) << 16);
  reinterpret_cast<u32x2*>(out)[i] = o;
}

// ---------------------------------------------------------------------------
// C[M,N] = A[M,K] * W[N,K]^T + bias   (A,W bf16; acc f32)
// block = 256 threads (8 waves), tile 128(M) x 128(N); wave tile 32x64.
// mode 0: store bf16 row-major [M,N]
// mode 1: store f32  row-major [M,N]
// mode 2: store bf16 transposed per head: Vt[b][h][d][s]
// ---------------------------------------------------------------------------
__global__ __launch_bounds__(256)
void gemm_bf16_wmma_kernel(const unsigned short* __restrict__ A,
                           const unsigned short* __restrict__ W,
                           const float* __restrict__ bias,
                           void* __restrict__ out, int mode) {
  constexpr int K = Hc, N = Hc;
  const int mBase = blockIdx.x * 128;
  const int nBase = blockIdx.y * 128;
  const int tid   = threadIdx.x;
  const int wid   = tid >> 5;
  const int lane  = tid & 31;
  const int lm    = lane & 15;
  const int khalf = lane >> 4;       // 0: K 0..7/16..23 halves, 1: 8..15/24..31
  const int wm    = wid >> 1;        // 0..3 -> M offset
  const int wn    = wid & 1;         // 0..1 -> N offset

  f32x8 acc[2][4] = {};

  const unsigned short* arow[2];
  const unsigned short* wrow[4];
#pragma unroll
  for (int mt = 0; mt < 2; ++mt)
    arow[mt] = A + (size_t)(mBase + wm * 32 + mt * 16 + lm) * K;
#pragma unroll
  for (int nt = 0; nt < 4; ++nt)
    wrow[nt] = W + (size_t)(nBase + wn * 64 + nt * 16 + lm) * K;

  for (int k0 = 0; k0 < K; k0 += 32) {
    Frag af[2], bf[4];
#pragma unroll
    for (int mt = 0; mt < 2; ++mt) {
      af[mt].u[0] = *reinterpret_cast<const u32x4*>(arow[mt] + k0 + khalf * 8);
      af[mt].u[1] = *reinterpret_cast<const u32x4*>(arow[mt] + k0 + 16 + khalf * 8);
    }
#pragma unroll
    for (int nt = 0; nt < 4; ++nt) {
      bf[nt].u[0] = *reinterpret_cast<const u32x4*>(wrow[nt] + k0 + khalf * 8);
      bf[nt].u[1] = *reinterpret_cast<const u32x4*>(wrow[nt] + k0 + 16 + khalf * 8);
      __builtin_prefetch(wrow[nt] + k0 + 32, 0, 0);   // global_prefetch_b8
    }
#pragma unroll
    for (int mt = 0; mt < 2; ++mt)
#pragma unroll
      for (int nt = 0; nt < 4; ++nt)
        acc[mt][nt] = wmma_bf16(af[mt].b, bf[nt].b, acc[mt][nt]);
  }

  // Epilogue: C/D layout -> (m = r + 8*khalf, n = lm) per 16x16 tile
#pragma unroll
  for (int mt = 0; mt < 2; ++mt) {
#pragma unroll
    for (int nt = 0; nt < 4; ++nt) {
      const int n = nBase + wn * 64 + nt * 16 + lm;
      const float bn = bias[n];
#pragma unroll
      for (int r = 0; r < 8; ++r) {
        const int m = mBase + wm * 32 + mt * 16 + r + khalf * 8;
        const float v = acc[mt][nt][r] + bn;
        if (mode == 0) {
          ((unsigned short*)out)[(size_t)m * N + n] = f2bf(v);
        } else if (mode == 1) {
          ((float*)out)[(size_t)m * N + n] = v;
        } else {                                   // V: [b][h][d][s]
          const int bb = m >> 11, s = m & (Sc - 1);
          const int h = n >> 6,  d = n & 63;
          ((unsigned short*)out)[(((size_t)(bb * NHc + h)) * HDc + d) * Sc + s] =
              f2bf(v);
        }
      }
    }
  }
}

// ---------------------------------------------------------------------------
// Fused attention core for one (b, h, 16-query stripe):
//   energy = (Q Kt)/8 -> LDS; mask+softmax in LDS; write attention to HBM;
//   ctx = P V via WMMA with cross-wave ds_add_f32 reduction.
// LDS: scores 16*2048 f32 (128K) + P bf16 (64K) + ctx acc (4K) = 196.6 KB
// ---------------------------------------------------------------------------
__global__ __launch_bounds__(256)
void attention_kernel(const unsigned short* __restrict__ Qp,
                      const unsigned short* __restrict__ Kp,
                      const unsigned short* __restrict__ Vt,
                      const int* __restrict__ mask,
                      float* __restrict__ attn,
                      unsigned short* __restrict__ ctx) {
  extern __shared__ __align__(16) char smem[];
  float*          sc   = reinterpret_cast<float*>(smem);                    // [16][S]
  unsigned short* pb   = reinterpret_cast<unsigned short*>(smem + 16 * Sc * 4);
  float*          cacc = reinterpret_cast<float*>(smem + 16 * Sc * 4 + 16 * Sc * 2);

  const int qt = blockIdx.x, h = blockIdx.y, b = blockIdx.z;
  const int qBase = qt * 16;
  const int tid = threadIdx.x, wid = tid >> 5, lane = tid & 31;
  const int lm = lane & 15, khalf = lane >> 4;

  // ---- Phase 1: energy tiles (each wave covers 256 keys = 16 WMMA tiles) ----
  const unsigned short* qrow =
      Qp + (size_t)(b * Sc + qBase + lm) * Hc + h * HDc;
  Frag qf[2];
#pragma unroll
  for (int kd = 0; kd < 2; ++kd) {
    qf[kd].u[0] = *reinterpret_cast<const u32x4*>(qrow + kd * 32 + khalf * 8);
    qf[kd].u[1] = *reinterpret_cast<const u32x4*>(qrow + kd * 32 + 16 + khalf * 8);
  }
  for (int t = 0; t < 16; ++t) {
    const int kcol = wid * 256 + t * 16;
    const unsigned short* krow =
        Kp + (size_t)(b * Sc + kcol + lm) * Hc + h * HDc;
    f32x8 e = {};
#pragma unroll
    for (int kd = 0; kd < 2; ++kd) {
      Frag kf;
      kf.u[0] = *reinterpret_cast<const u32x4*>(krow + kd * 32 + khalf * 8);
      kf.u[1] = *reinterpret_cast<const u32x4*>(krow + kd * 32 + 16 + khalf * 8);
      e = wmma_bf16(qf[kd].b, kf.b, e);
    }
#pragma unroll
    for (int r = 0; r < 8; ++r)
      sc[(r + khalf * 8) * Sc + kcol + lm] = e[r] * kInvScale;
  }
  __syncthreads();

  // ---- Phase 2: mask + softmax per row (wave -> 2 rows), emit attention ----
#pragma unroll
  for (int rr = 0; rr < 2; ++rr) {
    const int row = wid * 2 + rr;
    const int q = qBase + row;
    const int* mrow = mask + ((size_t)b * Sc + q) * Sc;
    float* srow = sc + row * Sc;
    float mx = -3.0e38f;
    for (int c = lane; c < Sc; c += 32) {
      float v = (mrow[c] != 0) ? srow[c] : kNeg;
      srow[c] = v;
      mx = fmaxf(mx, v);
    }
#pragma unroll
    for (int off = 16; off > 0; off >>= 1)
      mx = fmaxf(mx, __shfl_xor(mx, off, 32));
    float sum = 0.f;
    for (int c = lane; c < Sc; c += 32) {
      float e = __expf(srow[c] - mx);
      srow[c] = e;
      sum += e;
    }
#pragma unroll
    for (int off = 16; off > 0; off >>= 1)
      sum += __shfl_xor(sum, off, 32);
    const float inv = 1.0f / sum;
    float* arow = attn + (((size_t)(b * NHc + h) * Sc) + q) * Sc;
    unsigned short* prow = pb + row * Sc;
    for (int c = lane; c < Sc; c += 32) {
      const float p = srow[c] * inv;
      arow[c] = p;              // attention output (the HBM-bound write)
      prow[c] = f2bf(p);        // bf16 P for the PV matmul
    }
  }
  for (int i = tid; i < 16 * HDc; i += 256) cacc[i] = 0.f;
  __syncthreads();

  // ---- Phase 3: ctx = P V   (wave wid reduces K slice [wid*256, +256)) ----
  f32x8 oacc[4] = {};
  const size_t vbase = (size_t)(b * NHc + h) * HDc * Sc;
  for (int kk = 0; kk < 8; ++kk) {
    const int k0 = wid * 256 + kk * 32;
    Frag af;
    const unsigned short* prow = pb + lm * Sc + k0;
    af.u[0] = *reinterpret_cast<const u32x4*>(prow + khalf * 8);
    af.u[1] = *reinterpret_cast<const u32x4*>(prow + 16 + khalf * 8);
#pragma unroll
    for (int nt = 0; nt < 4; ++nt) {
      Frag vf;
      const unsigned short* vrow =
          Vt + vbase + (size_t)(nt * 16 + lm) * Sc + k0;
      vf.u[0] = *reinterpret_cast<const u32x4*>(vrow + khalf * 8);
      vf.u[1] = *reinterpret_cast<const u32x4*>(vrow + 16 + khalf * 8);
      oacc[nt] = wmma_bf16(af.b, vf.b, oacc[nt]);
    }
  }
#pragma unroll
  for (int nt = 0; nt < 4; ++nt)
#pragma unroll
    for (int r = 0; r < 8; ++r)
      atomicAdd(&cacc[(r + khalf * 8) * HDc + nt * 16 + lm], oacc[nt][r]);
  __syncthreads();

  for (int i = tid; i < 16 * HDc; i += 256) {
    const int row = i >> 6, col = i & 63;
    ctx[(size_t)(b * Sc + qBase + row) * Hc + h * HDc + col] = f2bf(cacc[i]);
  }
}

// ---------------------------------------------------------------------------
extern "C" void kernel_launch(void* const* d_in, const int* in_sizes, int n_in,
                              void* d_out, int out_size, void* d_ws,
                              size_t ws_size, hipStream_t stream) {
  (void)in_sizes; (void)n_in; (void)out_size; (void)ws_size;

  const float* q    = (const float*)d_in[0];
  const float* k    = (const float*)d_in[1];
  const float* v    = (const float*)d_in[2];
  const int*   mask = (const int*)d_in[3];
  const float* Wq = (const float*)d_in[4];  const float* bq = (const float*)d_in[5];
  const float* Wk = (const float*)d_in[6];  const float* bk = (const float*)d_in[7];
  const float* Wv = (const float*)d_in[8];  const float* bv = (const float*)d_in[9];
  const float* Wo = (const float*)d_in[10]; const float* bo = (const float*)d_in[11];

  char* ws = (char*)d_ws;
  const size_t nIn = (size_t)Mtot * Hc;   // 8,388,608 elements
  const size_t nW  = (size_t)Hc * Hc;     // 1,048,576 elements
  size_t off = 0;
  unsigned short* inQ = (unsigned short*)(ws + off); off += nIn * 2;
  unsigned short* inK = (unsigned short*)(ws + off); off += nIn * 2;
  unsigned short* inV = (unsigned short*)(ws + off); off += nIn * 2;
  unsigned short* wqb = (unsigned short*)(ws + off); off += nW * 2;
  unsigned short* wkb = (unsigned short*)(ws + off); off += nW * 2;
  unsigned short* wvb = (unsigned short*)(ws + off); off += nW * 2;
  unsigned short* wob = (unsigned short*)(ws + off); off += nW * 2;
  unsigned short* Qp  = (unsigned short*)(ws + off); off += nIn * 2;
  unsigned short* Kp  = (unsigned short*)(ws + off); off += nIn * 2;
  unsigned short* Vt  = (unsigned short*)(ws + off); off += nIn * 2;
  unsigned short* Ctx = (unsigned short*)(ws + off); off += nIn * 2;

  dim3 blk(256);
  const int n4in = (int)(nIn / 4), n4w = (int)(nW / 4);
  cvt_bf16_kernel<<<n4in / 256, blk, 0, stream>>>(q,  inQ, n4in);
  cvt_bf16_kernel<<<n4in / 256, blk, 0, stream>>>(k,  inK, n4in);
  cvt_bf16_kernel<<<n4in / 256, blk, 0, stream>>>(v,  inV, n4in);
  cvt_bf16_kernel<<<n4w / 256,  blk, 0, stream>>>(Wq, wqb, n4w);
  cvt_bf16_kernel<<<n4w / 256,  blk, 0, stream>>>(Wk, wkb, n4w);
  cvt_bf16_kernel<<<n4w / 256,  blk, 0, stream>>>(Wv, wvb, n4w);
  cvt_bf16_kernel<<<n4w / 256,  blk, 0, stream>>>(Wo, wob, n4w);

  dim3 ggrid(Mtot / 128, Hc / 128);    // 64 x 8 workgroups
  gemm_bf16_wmma_kernel<<<ggrid, blk, 0, stream>>>(inQ, wqb, bq, Qp, 0);
  gemm_bf16_wmma_kernel<<<ggrid, blk, 0, stream>>>(inK, wkb, bk, Kp, 0);
  gemm_bf16_wmma_kernel<<<ggrid, blk, 0, stream>>>(inV, wvb, bv, Vt, 2);

  float* outp = (float*)d_out;
  float* attn = outp + nIn;
  const size_t smemBytes = 16 * Sc * 4 + 16 * Sc * 2 + 16 * HDc * 4; // 200704
  attention_kernel<<<dim3(Sc / 16, NHc, Bc), blk, smemBytes, stream>>>(
      Qp, Kp, Vt, mask, attn, Ctx);

  gemm_bf16_wmma_kernel<<<ggrid, blk, 0, stream>>>(Ctx, wob, bo, (void*)outp, 1);
}
